// ConvLSTM_3942779977788
// MI455X (gfx1250) — matrix operand order repaired and checked by reading, not compiled, not run on hardware
//
#include <hip/hip_runtime.h>

typedef __attribute__((ext_vector_type(16))) __bf16 v16bf;
typedef __attribute__((ext_vector_type(8)))  float  v8f;

#define BB   32
#define CIN  32
#define HC   64
#define HH   64
#define WW   64
#define KXD  288        // CIN*9 (weight row length, x-convs)
#define KHD  576        // HC*9  (weight row length, h-convs)
#define KTOT 864        // 9 taps * 96 channels
#define LDR  104        // LDS row stride (elements): 208B = 13*16B, keeps b128 frags aligned

__device__ __forceinline__ void st_pk_bf16(__bf16* dst, float a, float b) {
    union { __bf16 h[2]; unsigned u; } pk;
    pk.h[0] = (__bf16)a;
    pk.h[1] = (__bf16)b;
    *(unsigned*)dst = pk.u;          // -> v_cvt_pk_bf16_f32 + ds_store_b32
}

__device__ __forceinline__ unsigned lds_addr32(const void* p) {
    // generic LDS pointer: low 32 bits == LDS byte offset (aperture rule, ISA 10.2)
    return (unsigned)(unsigned long long)p;
}

// ---------------------------------------------------------------------------
// Prep kernel: repack all 8 weight tensors into bf16, GEMM K-order k = t*96+c.
// wpk[(g*64+n)*864 + t*96 + c], c<32 -> x-conv channel c, else h-conv channel c-32.
// ---------------------------------------------------------------------------
__global__ __launch_bounds__(96)
void repack_weights(const float* __restrict__ wxi, const float* __restrict__ wxf,
                    const float* __restrict__ wxo, const float* __restrict__ wxg,
                    const float* __restrict__ whi, const float* __restrict__ whf,
                    const float* __restrict__ who, const float* __restrict__ whg,
                    __bf16* __restrict__ wpk)
{
    const int n = blockIdx.x;        // 0..63
    const int g = blockIdx.y;        // 0..3
    const int c = threadIdx.x;       // 0..95
    const float* wxp[4] = {wxi, wxf, wxo, wxg};
    const float* whp[4] = {whi, whf, who, whg};
    const float* wx = wxp[g];
    const float* wh = whp[g];
    #pragma unroll
    for (int t = 0; t < 9; ++t) {
        float v = (c < 32) ? wx[n * KXD + c * 9 + t]
                           : wh[n * KHD + (c - 32) * 9 + t];
        wpk[(g * 64 + n) * KTOT + t * 96 + c] = (__bf16)v;
    }
}

// Issue the async global->LDS copy of one tap's B tile (128 rows x 192B).
__device__ __forceinline__ void issue_async_B(const __bf16* wpk, __bf16* dstBuf,
                                              int nBase, int t, int tid)
{
    #pragma unroll
    for (int e = 0; e < 6; ++e) {
        int pid = tid + 256 * e;          // 0..1535 = 128 rows * 12 b128-chunks
        int j   = pid % 12;
        int ng  = pid / 12;               // gate*32 + n_local
        int gr  = (ng >> 5) * 64 + nBase + (ng & 31);
        unsigned goff = (unsigned)((gr * KTOT + t * 96) * 2 + j * 16);
        unsigned la   = lds_addr32(&dstBuf[ng * LDR + j * 8]);
        asm volatile("global_load_async_to_lds_b128 %0, %1, %2"
                     :: "v"(la), "v"(goff), "s"(wpk) : "memory");
    }
}

// ---------------------------------------------------------------------------
// Fast kernel: raw input rows staged once (taps read shifted views), B tiles
// double-buffered via async global->LDS from repacked bf16 weights.
// ---------------------------------------------------------------------------
__global__ __launch_bounds__(256)
void convlstm_fast(const float* __restrict__ X,   const float* __restrict__ Hst,
                   const float* __restrict__ Cst, const __bf16* __restrict__ wpk,
                   const float* __restrict__ bxi, const float* __restrict__ bxf,
                   const float* __restrict__ bxo, const float* __restrict__ bxg,
                   const float* __restrict__ bhi, const float* __restrict__ bhf,
                   const float* __restrict__ bho, const float* __restrict__ bhg,
                   float* __restrict__ outH, float* __restrict__ outC)
{
    // Raw activations: 3 rows (y-1,y,y+1) x 66 px (x = -1..64, zero-padded) x 96 ch.
    __shared__ __align__(16) __bf16 ldsRaw[3 * 66 * LDR];     // 41184 B
    __shared__ __align__(16) __bf16 ldsB[2][128 * LDR];       // 2 x 26624 B

    const int tid  = threadIdx.x;
    const int lane = tid & 31;
    const int wid  = tid >> 5;
    const int hb   = (lane >> 4) & 1;
    const int l16  = lane & 15;

    const int brow  = blockIdx.x;
    const int b     = brow >> 6;
    const int y     = brow & 63;
    const int nBase = blockIdx.y * 32;

    const int m_sub = wid & 3;
    const int n_sub = wid >> 2;

    const int ch   = nBase + n_sub * 16 + l16;
    const int xb   = m_sub * 16 + hb * 8;
    const int coff = (((b * HC + ch) * HH) + y) * WW + xb;

    __builtin_prefetch(Cst + coff, 0, 0);

    // Kick off tap-0 B tile copy immediately.
    issue_async_B(wpk, ldsB[0], nBase, 0, tid);

    // Zero the pad columns (ix = -1 and ix = 64): 3 rows x 2 cols x 48 pairs.
    #pragma unroll
    for (int e = 0; e < 2; ++e) {
        int q = tid + 256 * e;
        if (q < 288) {
            int c2 = q % 48;
            int rs = q / 48;               // 0..5
            int r  = rs >> 1;
            int s  = (rs & 1) ? 65 : 0;
            st_pk_bf16(&ldsRaw[(r * 66 + s) * LDR + 2 * c2], 0.0f, 0.0f);
        }
    }

    // Stage raw rows once: 3 rows x 64 ix x 48 channel-pairs (36 pairs/thread).
    #pragma unroll
    for (int e = 0; e < 36; ++e) {
        int pid  = tid + 256 * e;          // 0..9215
        int ix   = pid & 63;
        int rest = pid >> 6;               // 0..143
        int cp   = rest % 48;
        int r    = rest / 48;              // 0..2
        int ry   = y + r - 1;
        float v0 = 0.0f, v1 = 0.0f;
        if ((unsigned)ry < 64u) {
            if (cp < 16) {                 // x channels
                int base = (((b * CIN) + 2 * cp) * HH + ry) * WW + ix;
                v0 = X[base];
                v1 = X[base + HH * WW];
            } else {                       // h channels
                int hc = 2 * cp - 32;
                int base = (((b * HC) + hc) * HH + ry) * WW + ix;
                v0 = Hst[base];
                v1 = Hst[base + HH * WW];
            }
        }
        st_pk_bf16(&ldsRaw[(r * 66 + ix + 1) * LDR + 2 * cp], v0, v1);
    }

    v8f acc[4] = {};
    const int pPix = m_sub * 16 + l16;     // pixel row (M) owned by this lane

    #pragma unroll 1
    for (int t = 0; t < 9; ++t) {
        asm volatile("s_wait_asynccnt 0x0" ::: "memory");  // own B[t] chunks landed
        __syncthreads();                                   // everyone's landed; prior reads done
        if (t < 8)
            issue_async_B(wpk, ldsB[(t + 1) & 1], nBase, t + 1, tid);

        const int kh = t / 3, kw = t % 3;
        const __bf16* bufB = ldsB[t & 1];
        const __bf16* arow = ldsRaw + (kh * 66 + pPix + kw) * LDR;  // shifted raw view

        #pragma unroll
        for (int kcc = 0; kcc < 3; ++kcc) {
            const int c0 = kcc * 32;
            union { uint4 q[2]; v16bf v; } af;
            af.q[0] = *(const uint4*)(arow + c0 + hb * 8);
            af.q[1] = *(const uint4*)(arow + c0 + 16 + hb * 8);

            #pragma unroll
            for (int g = 0; g < 4; ++g) {
                union { uint4 q[2]; v16bf v; } bfr;
                const __bf16* bbase = bufB + (g * 32 + n_sub * 16 + l16) * LDR + c0 + hb * 8;
                bfr.q[0] = *(const uint4*)(bbase);
                bfr.q[1] = *(const uint4*)(bbase + 16);
                acc[g] = __builtin_amdgcn_wmma_f32_16x16x32_bf16(
                             false, af.v, false, bfr.v, (short)0, acc[g], false, false);
            }
        }
    }

    const float bi = bxi[ch] + bhi[ch];
    const float bf = bxf[ch] + bhf[ch];
    const float bo = bxo[ch] + bho[ch];
    const float bg = bxg[ch] + bhg[ch];

    #pragma unroll
    for (int r = 0; r < 8; ++r) {
        float zi = acc[0][r] + bi;
        float zf = acc[1][r] + bf;
        float zo = acc[2][r] + bo;
        float zg = acc[3][r] + bg;
        float iv = 1.0f / (1.0f + __expf(-zi));
        float fv = 1.0f / (1.0f + __expf(-zf));
        float ov = 1.0f / (1.0f + __expf(-zo));
        float gv = tanhf(zg);
        float cold = Cst[coff + r];
        float cnew = fv * cold + iv * gv;
        float hnew = ov * tanhf(cnew);
        outH[coff + r] = hnew;
        outC[coff + r] = cnew;
    }
}

// ---------------------------------------------------------------------------
// Fallback kernel (self-staging): used only if the workspace
// cannot hold the repacked weights.
// ---------------------------------------------------------------------------
__global__ __launch_bounds__(256)
void convlstm_wmma_bf16(const float* __restrict__ X,   const float* __restrict__ Hst,
                        const float* __restrict__ Cst,
                        const float* __restrict__ wxi, const float* __restrict__ bxi,
                        const float* __restrict__ wxf, const float* __restrict__ bxf,
                        const float* __restrict__ wxo, const float* __restrict__ bxo,
                        const float* __restrict__ wxg, const float* __restrict__ bxg,
                        const float* __restrict__ whi, const float* __restrict__ bhi,
                        const float* __restrict__ whf, const float* __restrict__ bhf,
                        const float* __restrict__ who, const float* __restrict__ bho,
                        const float* __restrict__ whg, const float* __restrict__ bhg,
                        float* __restrict__ outH, float* __restrict__ outC)
{
    __shared__ __align__(16) __bf16 ldsA[64 * LDR];
    __shared__ __align__(16) __bf16 ldsBf[128 * LDR];

    const int tid  = threadIdx.x;
    const int lane = tid & 31;
    const int wid  = tid >> 5;
    const int hb   = (lane >> 4) & 1;
    const int l16  = lane & 15;

    const int brow  = blockIdx.x;
    const int b     = brow >> 6;
    const int y     = brow & 63;
    const int nBase = blockIdx.y * 32;

    const float* wxp[4] = {wxi, wxf, wxo, wxg};
    const float* whp[4] = {whi, whf, who, whg};

    const int m_sub = wid & 3;
    const int n_sub = wid >> 2;

    const int ch   = nBase + n_sub * 16 + l16;
    const int xb   = m_sub * 16 + hb * 8;
    const int coff = (((b * HC + ch) * HH) + y) * WW + xb;

    __builtin_prefetch(Cst + coff, 0, 0);

    v8f acc[4] = {};

    #pragma unroll 1
    for (int t = 0; t < 9; ++t) {
        const int kh = t / 3, kw = t % 3;
        const int iy = y + kh - 1;
        const bool iyok = (unsigned)iy < 64u;

        #pragma unroll
        for (int e = 0; e < 4; ++e) {
            int pid = tid + 256 * e;
            int p   = pid & 63;
            int cp  = pid >> 6;
            int ix  = p + kw - 1;
            float v0 = 0.0f, v1 = 0.0f;
            if (iyok && (unsigned)ix < 64u) {
                int base = (((b * CIN) + 2 * cp) * HH + iy) * WW + ix;
                v0 = X[base];
                v1 = X[base + HH * WW];
            }
            st_pk_bf16(&ldsA[p * LDR + 2 * cp], v0, v1);
        }
        #pragma unroll
        for (int e = 4; e < 12; ++e) {
            int pid = tid + 256 * e;
            int p   = pid & 63;
            int cp  = pid >> 6;
            int hc  = 2 * cp - 32;
            int ix  = p + kw - 1;
            float v0 = 0.0f, v1 = 0.0f;
            if (iyok && (unsigned)ix < 64u) {
                int base = (((b * HC) + hc) * HH + iy) * WW + ix;
                v0 = Hst[base];
                v1 = Hst[base + HH * WW];
            }
            st_pk_bf16(&ldsA[p * LDR + 2 * cp], v0, v1);
        }
        #pragma unroll
        for (int e = 0; e < 24; ++e) {
            int pid = tid + 256 * e;
            int cp  = pid % 48;
            int ng  = pid / 48;
            int g   = ng >> 5;
            int nl  = ng & 31;
            int n   = nBase + nl;
            int c   = 2 * cp;
            float v0, v1;
            if (c < 32) {
                int wi = n * KXD + c * 9 + t;
                v0 = wxp[g][wi];
                v1 = wxp[g][wi + 9];
            } else {
                int wi = n * KHD + (c - 32) * 9 + t;
                v0 = whp[g][wi];
                v1 = whp[g][wi + 9];
            }
            st_pk_bf16(&ldsBf[ng * LDR + c], v0, v1);
        }
        __syncthreads();

        #pragma unroll
        for (int kcc = 0; kcc < 3; ++kcc) {
            const int c0 = kcc * 32;
            union { uint4 q[2]; v16bf v; } af;
            const __bf16* abase = ldsA + (m_sub * 16 + l16) * LDR + c0;
            af.q[0] = *(const uint4*)(abase + hb * 8);
            af.q[1] = *(const uint4*)(abase + 16 + hb * 8);

            #pragma unroll
            for (int g = 0; g < 4; ++g) {
                union { uint4 q[2]; v16bf v; } bfr;
                const __bf16* bbase = ldsBf + (g * 32 + n_sub * 16 + l16) * LDR + c0 + hb * 8;
                bfr.q[0] = *(const uint4*)(bbase);
                bfr.q[1] = *(const uint4*)(bbase + 16);
                acc[g] = __builtin_amdgcn_wmma_f32_16x16x32_bf16(
                             false, af.v, false, bfr.v, (short)0, acc[g], false, false);
            }
        }
        __syncthreads();
    }

    const float bi = bxi[ch] + bhi[ch];
    const float bf = bxf[ch] + bhf[ch];
    const float bo = bxo[ch] + bho[ch];
    const float bg = bxg[ch] + bhg[ch];

    #pragma unroll
    for (int r = 0; r < 8; ++r) {
        float zi = acc[0][r] + bi;
        float zf = acc[1][r] + bf;
        float zo = acc[2][r] + bo;
        float zg = acc[3][r] + bg;
        float iv = 1.0f / (1.0f + __expf(-zi));
        float fv = 1.0f / (1.0f + __expf(-zf));
        float ov = 1.0f / (1.0f + __expf(-zo));
        float gv = tanhf(zg);
        float cold = Cst[coff + r];
        float cnew = fv * cold + iv * gv;
        float hnew = ov * tanhf(cnew);
        outH[coff + r] = hnew;
        outC[coff + r] = cnew;
    }
}

extern "C" void kernel_launch(void* const* d_in, const int* in_sizes, int n_in,
                              void* d_out, int out_size, void* d_ws, size_t ws_size,
                              hipStream_t stream) {
    const float* X   = (const float*)d_in[0];
    const float* Hst = (const float*)d_in[1];
    const float* Cst = (const float*)d_in[2];
    const float* wxi = (const float*)d_in[3];
    const float* bxi = (const float*)d_in[4];
    const float* wxf = (const float*)d_in[5];
    const float* bxf = (const float*)d_in[6];
    const float* wxo = (const float*)d_in[7];
    const float* bxo = (const float*)d_in[8];
    const float* wxg = (const float*)d_in[9];
    const float* bxg = (const float*)d_in[10];
    const float* whi = (const float*)d_in[11];
    const float* bhi = (const float*)d_in[12];
    const float* whf = (const float*)d_in[13];
    const float* bhf = (const float*)d_in[14];
    const float* who = (const float*)d_in[15];
    const float* bho = (const float*)d_in[16];
    const float* whg = (const float*)d_in[17];
    const float* bhg = (const float*)d_in[18];

    float* outH = (float*)d_out;
    float* outC = outH + (size_t)BB * HC * HH * WW;

    dim3 grid(2048, 2, 1);
    const size_t wsNeeded = (size_t)4 * 64 * KTOT * 2;   // 442368 B repacked weights

    if (ws_size >= wsNeeded) {
        __bf16* wpk = (__bf16*)d_ws;
        repack_weights<<<dim3(64, 4, 1), 96, 0, stream>>>(
            wxi, wxf, wxo, wxg, whi, whf, who, whg, wpk);
        convlstm_fast<<<grid, 256, 0, stream>>>(
            X, Hst, Cst, wpk,
            bxi, bxf, bxo, bxg, bhi, bhf, bho, bhg,
            outH, outC);
    } else {
        convlstm_wmma_bf16<<<grid, 256, 0, stream>>>(
            X, Hst, Cst,
            wxi, bxi, wxf, bxf, wxo, bxo, wxg, bxg,
            whi, bhi, whf, bhf, who, bho, whg, bhg,
            outH, outC);
    }
}
